// SAModule_Pointcnn_59923383714428
// MI455X (gfx1250) — matrix operand order repaired and checked by reading, not compiled, not run on hardware
//
#include <hip/hip_runtime.h>
#include <hip/hip_bf16.h>
#include <math.h>

#define B_    8
#define NPER  16384
#define DIMP  6
#define KNN   16
#define CIN   64
#define HID   32
#define COUT  128
#define MS    4096
#define MT    (B_*MS)          /* 32768 */
#define CCAT  (CIN+HID)        /* 96  */
#define DMUL  2
#define KK    (KNN*KNN)        /* 256 */
#define EPSBN 1e-5f

typedef __attribute__((ext_vector_type(16))) _Float16 v16h;
typedef __attribute__((ext_vector_type(8)))  float    v8f;

union V16H { v16h v; _Float16 h[16]; };
union V8F  { v8f  v; float    f[8]; };

__device__ __forceinline__ float eluf(float x) { return x > 0.f ? x : (expf(x) - 1.f); }

__device__ __forceinline__ v8f wmma16x16x32(v16h a, v16h b, v8f c) {
    return __builtin_amdgcn_wmma_f32_16x16x32_f16(false, a, false, b, (short)0, c, false, false);
}
// A-fragment K index for 16-bit A 16x32 (ISA 7.12.2 table)
__device__ __forceinline__ int a_kidx(int e, int lane) {
    return ((e < 8) ? e : e + 8) + ((lane >> 4) << 3);
}

// ---------------- FPS: one workgroup per cloud ----------------
__global__ __launch_bounds__(1024) void fps_kernel(const float* __restrict__ pos,
                                                   int* __restrict__ sidx) {
    const int b = blockIdx.x;
    const int t = threadIdx.x;
    const float* pb = pos + (size_t)b * NPER * DIMP;
    __shared__ float wv[32];
    __shared__ int   wi[32];
    __shared__ float q[DIMP];

    float dmin[16];
    float q0[DIMP];
#pragma unroll
    for (int d = 0; d < DIMP; ++d) q0[d] = pb[d];
#pragma unroll
    for (int u = 0; u < 16; ++u) {
        int j = t + 1024 * u;
        const float* p = pb + (size_t)j * DIMP;
        float s = 0.f;
#pragma unroll
        for (int d = 0; d < DIMP; ++d) { float df = p[d] - q0[d]; s += df * df; }
        dmin[u] = s;
    }
    if (t == 0) sidx[b * MS] = 0;
    const int lane = t & 31, wid = t >> 5;

    for (int i = 1; i < MS; ++i) {
        float bv = -1.f; int bj = 0;
#pragma unroll
        for (int u = 0; u < 16; ++u) {
            int j = t + 1024 * u;
            if (dmin[u] > bv || (dmin[u] == bv && j < bj)) { bv = dmin[u]; bj = j; }
        }
#pragma unroll
        for (int off = 16; off > 0; off >>= 1) {
            float ov = __shfl_down(bv, off);
            int   oj = __shfl_down(bj, off);
            if (ov > bv || (ov == bv && oj < bj)) { bv = ov; bj = oj; }
        }
        if (lane == 0) { wv[wid] = bv; wi[wid] = bj; }
        __syncthreads();
        if (wid == 0) {
            float v2 = wv[lane]; int j2 = wi[lane];
#pragma unroll
            for (int off = 16; off > 0; off >>= 1) {
                float ov = __shfl_down(v2, off);
                int   oj = __shfl_down(j2, off);
                if (ov > v2 || (ov == v2 && oj < j2)) { v2 = ov; j2 = oj; }
            }
            if (lane == 0) {
                sidx[b * MS + i] = j2;
                const float* p = pb + (size_t)j2 * DIMP;
#pragma unroll
                for (int d = 0; d < DIMP; ++d) q[d] = p[d];
            }
        }
        __syncthreads();
        float qq[DIMP];
#pragma unroll
        for (int d = 0; d < DIMP; ++d) qq[d] = q[d];
#pragma unroll
        for (int u = 0; u < 16; ++u) {
            int j = t + 1024 * u;
            const float* p = pb + (size_t)j * DIMP;
            float s = 0.f;
#pragma unroll
            for (int d = 0; d < DIMP; ++d) { float df = p[d] - qq[d]; s += df * df; }
            dmin[u] = fminf(dmin[u], s);
        }
    }
}

// ---------------- gather sampled pos/features + direct outputs ----------------
__global__ __launch_bounds__(256) void gather_kernel(const float* __restrict__ pos,
                                                     const float* __restrict__ x,
                                                     const int* __restrict__ sidx,
                                                     float* __restrict__ pos_s,
                                                     float* __restrict__ x_s,
                                                     float* __restrict__ pos_out,
                                                     int* __restrict__ batch_out) {
    int i = blockIdx.x * 256 + threadIdx.x;
    if (i >= MT) return;
    int b = i / MS;
    int j = sidx[i];
    const float* ps = pos + ((size_t)b * NPER + j) * DIMP;
#pragma unroll
    for (int d = 0; d < DIMP; ++d) { float v = ps[d]; pos_s[(size_t)i * DIMP + d] = v; pos_out[(size_t)i * DIMP + d] = v; }
    const float* xs = x + ((size_t)b * NPER + j) * CIN;
    for (int c = 0; c < CIN; ++c) x_s[(size_t)i * CIN + c] = xs[c];
    batch_out[i] = b;
}

// ---------------- KNN among samples, LDS-tiled brute force ----------------
#define KTILE 2048
__global__ __launch_bounds__(256) void knn_kernel(const float* __restrict__ pos_s,
                                                  int* __restrict__ nidx) {
    __shared__ float sp[KTILE * DIMP];   // 48 KB
    int b = blockIdx.x / (MS / 256);
    int q = (blockIdx.x % (MS / 256)) * 256 + threadIdx.x;
    const float* pb = pos_s + (size_t)b * MS * DIMP;
    float qc[DIMP];
#pragma unroll
    for (int d = 0; d < DIMP; ++d) qc[d] = pb[(size_t)q * DIMP + d];
    float bd[KNN]; int bi[KNN];
#pragma unroll
    for (int s = 0; s < KNN; ++s) { bd[s] = 3.4e38f; bi[s] = 0; }
    for (int tile = 0; tile < MS / KTILE; ++tile) {
        for (int e = threadIdx.x; e < KTILE * DIMP; e += 256)
            sp[e] = pb[(size_t)tile * KTILE * DIMP + e];
        __syncthreads();
        for (int j = 0; j < KTILE; ++j) {
            float s = 0.f;
#pragma unroll
            for (int d = 0; d < DIMP; ++d) { float df = sp[j * DIMP + d] - qc[d]; s += df * df; }
            if (s < bd[KNN - 1]) {
                float dd = s; int jj = tile * KTILE + j;
#pragma unroll
                for (int k = 0; k < KNN; ++k) {
                    if (dd < bd[k]) { float td = bd[k]; int ti = bi[k]; bd[k] = dd; bi[k] = jj; dd = td; jj = ti; }
                }
            }
        }
        __syncthreads();
    }
    size_t base = ((size_t)b * MS + q) * KNN;
#pragma unroll
    for (int k = 0; k < KNN; ++k) nidx[base + k] = bi[k];
}

// ---------------- rel positions + mlp1 layer1 (6->32, elu) ----------------
__global__ __launch_bounds__(256) void rel_h1_kernel(const float* __restrict__ pos_s,
                                                     const int* __restrict__ nidx,
                                                     const float* __restrict__ w1,
                                                     const float* __restrict__ b1,
                                                     float* __restrict__ rel,
                                                     float* __restrict__ hA) {
    __shared__ float sw[DIMP * HID];
    __shared__ float sb[HID];
    for (int e = threadIdx.x; e < DIMP * HID; e += 256) sw[e] = w1[e];
    if (threadIdx.x < HID) sb[threadIdx.x] = b1[threadIdx.x];
    __syncthreads();
    int R = blockIdx.x * 256 + threadIdx.x;      // row in [0, MT*KNN)
    int n = R / KNN, k = R % KNN;
    int b = n / MS;
    int j = nidx[(size_t)n * KNN + k];
    float r[DIMP];
    const float* pj = pos_s + ((size_t)b * MS + j) * DIMP;
    const float* pm = pos_s + (size_t)n * DIMP;
#pragma unroll
    for (int d = 0; d < DIMP; ++d) { r[d] = pj[d] - pm[d]; rel[(size_t)R * DIMP + d] = r[d]; }
    for (int o = 0; o < HID; ++o) {
        float a = sb[o];
#pragma unroll
        for (int d = 0; d < DIMP; ++d) a += r[d] * sw[d * HID + o];
        hA[(size_t)R * HID + o] = eluf(a);
    }
}

// ---------------- BN stats / finalize / apply ----------------
__global__ __launch_bounds__(256) void stats_kernel(const float* __restrict__ src,
                                                    long long total, int cmask,
                                                    float* __restrict__ acc) {
    __shared__ float ls[512];
    for (int e = threadIdx.x; e < 512; e += 256) ls[e] = 0.f;
    __syncthreads();
    long long stride = (long long)gridDim.x * 256;
    for (long long i = (long long)blockIdx.x * 256 + threadIdx.x; i < total; i += stride) {
        float v = src[i];
        int c = (int)(i & cmask);
        atomicAdd(&ls[c], v);
        atomicAdd(&ls[256 + c], v * v);
    }
    __syncthreads();
    if (threadIdx.x <= (unsigned)cmask) {
        atomicAdd(&acc[threadIdx.x], ls[threadIdx.x]);
        atomicAdd(&acc[256 + threadIdx.x], ls[256 + threadIdx.x]);
    }
}

__global__ void bn_finalize_kernel(float* __restrict__ acc, float invN,
                                   const float* __restrict__ g, const float* __restrict__ be, int cols) {
    int c = threadIdx.x;
    if (c < cols) {
        float mu  = acc[c] * invN;
        float var = acc[256 + c] * invN - mu * mu;
        float a   = g[c] * rsqrtf(var + EPSBN);
        acc[512 + c] = a;
        acc[768 + c] = be[c] - a * mu;
    }
}

__global__ __launch_bounds__(256) void bn_apply_kernel(float* __restrict__ x, long long total,
                                                       int cmask, const float* __restrict__ acc) {
    long long stride = (long long)gridDim.x * 256;
    for (long long i = (long long)blockIdx.x * 256 + threadIdx.x; i < total; i += stride) {
        int c = (int)(i & cmask);
        x[i] = acc[512 + c] * x[i] + acc[768 + c];
    }
}

// ---------------- mlp1 layer2 via WMMA: hB = elu(BN1(hA) @ w2 + b2) ----------------
__global__ __launch_bounds__(128) void gemm_h2_kernel(const float* __restrict__ hA,
                                                      const float* __restrict__ w2,
                                                      const float* __restrict__ b2,
                                                      const float* __restrict__ acc0,
                                                      float* __restrict__ hB) {
    const int lane = threadIdx.x & 31;
    const int tile = blockIdx.x * 4 + (threadIdx.x >> 5);
    const int tn = tile & 1;          // 32/16 = 2 N tiles
    const int tm = tile >> 1;
    const int r0 = lane & 15;
    V16H A, Bm; V8F C;
#pragma unroll
    for (int e = 0; e < 8; ++e) C.f[e] = 0.f;
    const float* ar = hA + (size_t)(tm * 16 + r0) * HID;
#pragma unroll
    for (int e = 0; e < 16; ++e) {
        int k = a_kidx(e, lane);
        A.h[e] = (_Float16)(acc0[512 + k] * ar[k] + acc0[768 + k]);  // fused BN1
    }
    const float* br = w2 + tn * 16 + (lane & 15);
#pragma unroll
    for (int e = 0; e < 16; ++e) Bm.h[e] = (_Float16)br[(size_t)a_kidx(e, lane) * HID];
    C.v = wmma16x16x32(A.v, Bm.v, C.v);
    const int col  = tn * 16 + (lane & 15);
    const float bias = b2[col];
    const int rbase = (lane >> 4) << 3;
#pragma unroll
    for (int e = 0; e < 8; ++e) {
        int row = e + rbase;
        hB[(size_t)(tm * 16 + row) * HID + col] = eluf(C.f[e] + bias);
    }
}

// ---------------- mlp2 linear via WMMA: tA = elu(rel(Mt x 96) @ wl + bl) ----------------
__global__ __launch_bounds__(128) void gemm_t1_kernel(const float* __restrict__ rel,
                                                      const float* __restrict__ wl,
                                                      const float* __restrict__ bl,
                                                      float* __restrict__ tA) {
    const int lane = threadIdx.x & 31;
    const int tile = blockIdx.x * 4 + (threadIdx.x >> 5);
    const int tn = tile & 15;         // 256/16 = 16 N tiles
    const int tm = tile >> 4;
    const int r0 = lane & 15;
    V16H A, Bm; V8F C;
#pragma unroll
    for (int e = 0; e < 8; ++e) C.f[e] = 0.f;
    const float* ar = rel + (size_t)(tm * 16 + r0) * (KNN * DIMP);
#pragma unroll
    for (int kk = 0; kk < 96; kk += 32) {
#pragma unroll
        for (int e = 0; e < 16; ++e) A.h[e] = (_Float16)ar[kk + a_kidx(e, lane)];
        const float* br = wl + (size_t)kk * KK + tn * 16 + (lane & 15);
#pragma unroll
        for (int e = 0; e < 16; ++e) Bm.h[e] = (_Float16)br[(size_t)a_kidx(e, lane) * KK];
        C.v = wmma16x16x32(A.v, Bm.v, C.v);
    }
    const int col = tn * 16 + (lane & 15);
    const float bias = bl[col];
    const int rbase = (lane >> 4) << 3;
#pragma unroll
    for (int e = 0; e < 8; ++e) {
        int row = e + rbase;
        tA[(size_t)(tm * 16 + row) * KK + col] = eluf(C.f[e] + bias);
    }
}

// ---------------- grouped Conv1d(K, K*K, K, groups=K) ----------------
__global__ __launch_bounds__(256) void gconv_kernel(const float* __restrict__ tin,
                                                    const float* __restrict__ cw,
                                                    const float* __restrict__ cb,
                                                    float* __restrict__ tout, int do_elu) {
    __shared__ float scw[KNN * KNN * KNN];   // 16 KB
    for (int e = threadIdx.x; e < KNN * KNN * KNN; e += 256) scw[e] = cw[e];
    __syncthreads();
    int T = blockIdx.x * 256 + threadIdx.x;  // (n, g)
    int n = T >> 4, g = T & 15;
    float tv[KNN];
    const float* tr = tin + (size_t)n * KK + g * KNN;
#pragma unroll
    for (int t = 0; t < KNN; ++t) tv[t] = tr[t];
    float* orow = tout + (size_t)n * KK + g * KNN;
    for (int m = 0; m < KNN; ++m) {
        float a = cb[g * KNN + m];
#pragma unroll
        for (int t = 0; t < KNN; ++t) a += tv[t] * scw[(g * KNN + m) * KNN + t];
        orow[m] = do_elu ? eluf(a) : a;
    }
}

// ---------------- X-transform + depthwise conv -> yflat (f16) ----------------
__global__ __launch_bounds__(256) void xdw_kernel(const float* __restrict__ tA,
                                                  const float* __restrict__ hB,
                                                  const float* __restrict__ x_s,
                                                  const int* __restrict__ nidx,
                                                  const float* __restrict__ wd,
                                                  const float* __restrict__ bd,
                                                  _Float16* __restrict__ yh) {
    __shared__ float st[8 * KK];   // 8 centers per block
    int lane = threadIdx.x & 31, wave = threadIdx.x >> 5;
    int n = blockIdx.x * 8 + wave;
    const float* tr = tA + (size_t)n * KK;
    for (int e = lane; e < KK; e += 32) st[wave * KK + e] = tr[e];
    __syncthreads();
    const float* tl = st + wave * KK;
    int b = n / MS;
    for (int ci = 0; ci < 3; ++ci) {
        int c = lane + 32 * ci;
        float xs[KNN];
        if (c < HID) {
#pragma unroll
            for (int k = 0; k < KNN; ++k) xs[k] = hB[((size_t)n * KNN + k) * HID + c];
        } else {
            int cc = c - HID;
#pragma unroll
            for (int k = 0; k < KNN; ++k) {
                int j = nidx[(size_t)n * KNN + k];
                xs[k] = x_s[((size_t)b * MS + j) * CIN + cc];
            }
        }
        float y0 = 0.f, y1 = 0.f;
#pragma unroll
        for (int jj = 0; jj < KNN; ++jj) {
            float xt = 0.f;
#pragma unroll
            for (int k = 0; k < KNN; ++k) xt += xs[k] * tl[k * KNN + jj];
            y0 += xt * wd[(c * DMUL + 0) * KNN + jj];
            y1 += xt * wd[(c * DMUL + 1) * KNN + jj];
        }
        yh[(size_t)n * (CCAT * DMUL) + c * DMUL + 0] = (_Float16)(y0 + bd[c * DMUL + 0]);
        yh[(size_t)n * (CCAT * DMUL) + c * DMUL + 1] = (_Float16)(y1 + bd[c * DMUL + 1]);
    }
}

__global__ void cvt_wf_kernel(const float* __restrict__ wf, _Float16* __restrict__ wfh) {
    int i = blockIdx.x * 256 + threadIdx.x;
    if (i < CCAT * DMUL * COUT) wfh[i] = (_Float16)wf[i];
}

// ---------------- final GEMM via WMMA: out = yflat(Mt x 192) @ wf + bf ----------------
__global__ __launch_bounds__(128) void gemm_out_kernel(const _Float16* __restrict__ yh,
                                                       const _Float16* __restrict__ wfh,
                                                       const float* __restrict__ bf,
                                                       float* __restrict__ out) {
    const int lane = threadIdx.x & 31;
    const int tile = blockIdx.x * 4 + (threadIdx.x >> 5);
    const int tn = tile & 7;          // 128/16 = 8 N tiles
    const int tm = tile >> 3;
    const int r0 = lane & 15;
    V16H A, Bm; V8F C;
#pragma unroll
    for (int e = 0; e < 8; ++e) C.f[e] = 0.f;
    const _Float16* ar = yh + (size_t)(tm * 16 + r0) * (CCAT * DMUL);
#pragma unroll
    for (int kc = 0; kc < 6; ++kc) {
        int kk = kc * 32;
#pragma unroll
        for (int e = 0; e < 16; ++e) A.h[e] = ar[kk + a_kidx(e, lane)];
        const _Float16* br = wfh + (size_t)kk * COUT + tn * 16 + (lane & 15);
#pragma unroll
        for (int e = 0; e < 16; ++e) Bm.h[e] = br[(size_t)a_kidx(e, lane) * COUT];
        C.v = wmma16x16x32(A.v, Bm.v, C.v);
    }
    const int col = tn * 16 + (lane & 15);
    const float bias = bf[col];
    const int rbase = (lane >> 4) << 3;
#pragma unroll
    for (int e = 0; e < 8; ++e) {
        int row = e + rbase;
        out[(size_t)(tm * 16 + row) * COUT + col] = C.f[e] + bias;
    }
}

extern "C" void kernel_launch(void* const* d_in, const int* in_sizes, int n_in,
                              void* d_out, int out_size, void* d_ws, size_t ws_size,
                              hipStream_t stream) {
    (void)in_sizes; (void)n_in; (void)out_size; (void)ws_size;
    const float* x    = (const float*)d_in[0];
    const float* pos  = (const float*)d_in[1];
    const float* w1   = (const float*)d_in[3];
    const float* b1   = (const float*)d_in[4];
    const float* g1   = (const float*)d_in[5];
    const float* be1  = (const float*)d_in[6];
    const float* w2   = (const float*)d_in[7];
    const float* b2   = (const float*)d_in[8];
    const float* g2   = (const float*)d_in[9];
    const float* be2  = (const float*)d_in[10];
    const float* wl   = (const float*)d_in[11];
    const float* bl   = (const float*)d_in[12];
    const float* gl   = (const float*)d_in[13];
    const float* bel  = (const float*)d_in[14];
    const float* cw1  = (const float*)d_in[15];
    const float* cb1  = (const float*)d_in[16];
    const float* gc1  = (const float*)d_in[17];
    const float* bec1 = (const float*)d_in[18];
    const float* cw2  = (const float*)d_in[19];
    const float* cb2  = (const float*)d_in[20];
    const float* gc2  = (const float*)d_in[21];
    const float* bec2 = (const float*)d_in[22];
    const float* wd   = (const float*)d_in[23];
    const float* bd   = (const float*)d_in[24];
    const float* wf   = (const float*)d_in[25];
    const float* bf   = (const float*)d_in[26];

    char* ws = (char*)d_ws;
    size_t off = 0;
    auto alloc = [&](size_t bytes) { void* p = ws + off; off += (bytes + 255) & ~(size_t)255; return p; };
    int*      sidx  = (int*)alloc((size_t)MT * 4);
    int*      nidx  = (int*)alloc((size_t)MT * KNN * 4);
    float*    pos_s = (float*)alloc((size_t)MT * DIMP * 4);
    float*    x_s   = (float*)alloc((size_t)MT * CIN * 4);
    float*    rel   = (float*)alloc((size_t)MT * KNN * DIMP * 4);
    float*    hA    = (float*)alloc((size_t)MT * KNN * HID * 4);
    float*    hB    = (float*)alloc((size_t)MT * KNN * HID * 4);
    float*    tA    = (float*)alloc((size_t)MT * KK * 4);
    float*    tB    = (float*)alloc((size_t)MT * KK * 4);
    _Float16* yh    = (_Float16*)alloc((size_t)MT * CCAT * DMUL * 2);
    _Float16* wfh   = (_Float16*)alloc((size_t)CCAT * DMUL * COUT * 2);
    float*    stats = (float*)alloc(5 * 1024 * 4);

    float* outp      = (float*)d_out;
    float* pos_out   = outp + (size_t)MT * COUT;
    int*   batch_out = (int*)(pos_out + (size_t)MT * DIMP);

    (void)hipMemsetAsync(stats, 0, 5 * 1024 * 4, stream);

    const long long NH = (long long)MT * KNN * HID;   // mlp1 elements
    const long long NT = (long long)MT * KK;          // mlp2 elements
    const float invNh = 1.f / ((float)MT * KNN);
    const float invNt = 1.f / (float)MT;

    fps_kernel<<<B_, 1024, 0, stream>>>(pos, sidx);
    gather_kernel<<<(MT + 255) / 256, 256, 0, stream>>>(pos, x, sidx, pos_s, x_s, pos_out, batch_out);
    knn_kernel<<<B_ * (MS / 256), 256, 0, stream>>>(pos_s, nidx);
    rel_h1_kernel<<<MT * KNN / 256, 256, 0, stream>>>(pos_s, nidx, w1, b1, rel, hA);

    stats_kernel<<<512, 256, 0, stream>>>(hA, NH, HID - 1, stats + 0 * 1024);
    bn_finalize_kernel<<<1, 256, 0, stream>>>(stats + 0 * 1024, invNh, g1, be1, HID);
    gemm_h2_kernel<<<(MT * KNN / 16) * 2 / 4, 128, 0, stream>>>(hA, w2, b2, stats + 0 * 1024, hB);
    stats_kernel<<<512, 256, 0, stream>>>(hB, NH, HID - 1, stats + 1 * 1024);
    bn_finalize_kernel<<<1, 256, 0, stream>>>(stats + 1 * 1024, invNh, g2, be2, HID);
    bn_apply_kernel<<<1024, 256, 0, stream>>>(hB, NH, HID - 1, stats + 1 * 1024);

    gemm_t1_kernel<<<(MT / 16) * 16 / 4, 128, 0, stream>>>(rel, wl, bl, tA);
    stats_kernel<<<512, 256, 0, stream>>>(tA, NT, 255, stats + 2 * 1024);
    bn_finalize_kernel<<<1, 256, 0, stream>>>(stats + 2 * 1024, invNt, gl, bel, 256);
    bn_apply_kernel<<<1024, 256, 0, stream>>>(tA, NT, 255, stats + 2 * 1024);

    gconv_kernel<<<MT * KNN / 256, 256, 0, stream>>>(tA, cw1, cb1, tB, 1);
    stats_kernel<<<512, 256, 0, stream>>>(tB, NT, 255, stats + 3 * 1024);
    bn_finalize_kernel<<<1, 256, 0, stream>>>(stats + 3 * 1024, invNt, gc1, bec1, 256);
    bn_apply_kernel<<<1024, 256, 0, stream>>>(tB, NT, 255, stats + 3 * 1024);

    gconv_kernel<<<MT * KNN / 256, 256, 0, stream>>>(tB, cw2, cb2, tA, 0);
    stats_kernel<<<512, 256, 0, stream>>>(tA, NT, 255, stats + 4 * 1024);
    bn_finalize_kernel<<<1, 256, 0, stream>>>(stats + 4 * 1024, invNt, gc2, bec2, 256);
    bn_apply_kernel<<<1024, 256, 0, stream>>>(tA, NT, 255, stats + 4 * 1024);

    xdw_kernel<<<MT / 8, 256, 0, stream>>>(tA, hB, x_s, nidx, wd, bd, yh);
    cvt_wf_kernel<<<(CCAT * DMUL * COUT + 255) / 256, 256, 0, stream>>>(wf, wfh);
    gemm_out_kernel<<<(MT / 16) * 8 / 4, 128, 0, stream>>>(yh, wfh, bf, outp);
}
